// PointNet2Model_24781961298017
// MI455X (gfx1250) — matrix operand order repaired and checked by reading, not compiled, not run on hardware
//
#include <hip/hip_runtime.h>
#include <hip/hip_bf16.h>
#include <math.h>

// ---------------------------------------------------------------------------
// Types for CDNA5 WMMA
// ---------------------------------------------------------------------------
typedef _Float16 v4h  __attribute__((ext_vector_type(4)));
typedef _Float16 v8h  __attribute__((ext_vector_type(8)));
typedef _Float16 v16h __attribute__((ext_vector_type(16)));
typedef float    v4f  __attribute__((ext_vector_type(4)));
typedef float    v8f  __attribute__((ext_vector_type(8)));

#define BATCH 8
#define NPTS  8192

#define TM 256          // rows per block (32 per wave, 2 WMMA sub-tiles)
#define TN 64
#define TK 32
#define LDP (TK + 8)    // padded LDS row (halves); 40*2B=80B, 16B-aligned rows

static inline int ceil_div_i(int a, int b) { return (a + b - 1) / b; }
static inline int pad4(int c) { return (c + 3) & ~3; }

// ---------------------------------------------------------------------------
// GEMM + bias + folded-BN + ReLU, f16 WMMA with f32 accumulate.
// Y[rows, cout] = epilogue( X[rows, cin] @ W[cin, cout] ),  X has row stride ldx.
// mode 0: y = relu((x@w + b) * g*rsqrt(1+eps) + be)   mode 1: y = x@w + b
// 256x64 tile/block, 8 waves, each wave = 32 rows (2 A frags) x 64 cols
// (4 B frags, each reused by 2 WMMAs). Double-buffered LDS, vectorized
// (b128) global fetch with register prefetch, one barrier per K-step.
// ---------------------------------------------------------------------------
__global__ __launch_bounds__(256) void gemm_bn_relu_kernel(
    const float* __restrict__ X, int ldx,
    const float* __restrict__ W,
    const float* __restrict__ bias, const float* __restrict__ gamma,
    const float* __restrict__ beta, float* __restrict__ Y,
    int rows, int cin, int cout, int mode)
{
    __shared__ __align__(16) _Float16 As[2][TM][LDP];
    __shared__ __align__(16) _Float16 Bs[2][TN][LDP];

    const int tid   = threadIdx.x;
    const int row0  = blockIdx.x * TM;
    const int n0    = blockIdx.y * TN;
    const int wave  = tid >> 5;
    const int lane  = tid & 31;
    const int lhalf = lane & 15;
    const int hsel  = lane >> 4;   // 0: lanes 0-15, 1: lanes 16-31
    const bool cout4 = ((cout & 3) == 0);

    v4f aR[8];  // 8 vec4 slots of the 256x32 A tile per thread
    v4f bR[2];  // 2 vec4 slots of the 32x64 B tile per thread

    auto fetch = [&](int kt) {
        const int k0 = kt * TK;
#pragma unroll
        for (int j = 0; j < 8; ++j) {
            const int s  = tid + j * 256;      // 0..2047
            const int r  = s >> 3;             // 8 vec4 per row
            const int kq = (s & 7) << 2;
            const int gr = row0 + r;
            v4f v = {};
            if (gr < rows) {
                if (k0 + kq + 4 <= cin) {
                    v = *(const v4f*)(X + (size_t)gr * ldx + k0 + kq);
                } else {
                    const float* xr = X + (size_t)gr * ldx;
#pragma unroll
                    for (int e = 0; e < 4; ++e) {
                        const int gk = k0 + kq + e;
                        v[e] = (gk < cin) ? xr[gk] : 0.0f;
                    }
                }
            }
            aR[j] = v;
        }
#pragma unroll
        for (int j = 0; j < 2; ++j) {
            const int s  = tid + j * 256;      // 0..511
            const int k  = s >> 4;             // 16 vec4 per k-row
            const int n  = (s & 15) << 2;
            const int gk = k0 + k;
            v4f v = {};
            if (gk < cin) {
                if (cout4 && (n0 + n + 4 <= cout)) {
                    v = *(const v4f*)(W + (size_t)gk * cout + n0 + n);
                } else {
                    const float* wr = W + (size_t)gk * cout;
#pragma unroll
                    for (int e = 0; e < 4; ++e) {
                        const int gn = n0 + n + e;
                        v[e] = (gn < cout) ? wr[gn] : 0.0f;
                    }
                }
            }
            bR[j] = v;
        }
    };

    auto stage = [&](int buf) {
#pragma unroll
        for (int j = 0; j < 8; ++j) {
            const int s  = tid + j * 256;
            const int r  = s >> 3;
            const int kq = (s & 7) << 2;
            v4h h;
#pragma unroll
            for (int e = 0; e < 4; ++e) h[e] = (_Float16)aR[j][e];
            *(v4h*)&As[buf][r][kq] = h;   // 8B-aligned ds_store_b64
        }
#pragma unroll
        for (int j = 0; j < 2; ++j) {
            const int s = tid + j * 256;
            const int k = s >> 4;
            const int n = (s & 15) << 2;
#pragma unroll
            for (int e = 0; e < 4; ++e)
                Bs[buf][n + e][k] = (_Float16)bR[j][e];  // transpose into LDS
        }
    };

    v8f acc[2][4];
    v8f zero = {};
#pragma unroll
    for (int ms = 0; ms < 2; ++ms)
#pragma unroll
        for (int i = 0; i < 4; ++i) acc[ms][i] = zero;

    const int ksteps = (cin + TK - 1) / TK;
    int buf = 0;
    fetch(0);
    stage(0);
    __syncthreads();

    for (int kt = 0; kt < ksteps; ++kt) {
        const bool more = (kt + 1) < ksteps;
        if (more) fetch(kt + 1);   // prefetch next tile into registers

        // Two A fragments (16x32 f16 each): rows wave*32 .. wave*32+31
        union { v16h v; v8h h8[2]; } af[2];
#pragma unroll
        for (int ms = 0; ms < 2; ++ms) {
            const int arow = wave * 32 + ms * 16 + lhalf;
            af[ms].h8[0] = *(const v8h*)&As[buf][arow][hsel ? 8  : 0];
            af[ms].h8[1] = *(const v8h*)&As[buf][arow][hsel ? 24 : 16];
        }

#pragma unroll
        for (int nb = 0; nb < 4; ++nb) {
            const int bcol = nb * 16 + lhalf;
            union { v16h v; v8h h8[2]; } bf;
            bf.h8[0] = *(const v8h*)&Bs[buf][bcol][hsel ? 8  : 0];
            bf.h8[1] = *(const v8h*)&Bs[buf][bcol][hsel ? 24 : 16];
            acc[0][nb] = __builtin_amdgcn_wmma_f32_16x16x32_f16(
                false, af[0].v, false, bf.v, (short)0, acc[0][nb], false, false);
            acc[1][nb] = __builtin_amdgcn_wmma_f32_16x16x32_f16(
                false, af[1].v, false, bf.v, (short)0, acc[1][nb], false, false);
        }

        if (more) {
            stage(buf ^ 1);        // write the other buffer: no conflict with reads
            __syncthreads();       // one barrier per K-step
            buf ^= 1;
        }
    }

    // Epilogue: C/D layout -> VGPR r: lanes0-15 M=r, lanes16-31 M=r+8; N=lane&15
    const float bnrs = rsqrtf(1.0f + 1e-5f);
#pragma unroll
    for (int ms = 0; ms < 2; ++ms) {
#pragma unroll
        for (int nb = 0; nb < 4; ++nb) {
            union { v8f v; float f[8]; } d;
            d.v = acc[ms][nb];
            const int gc = n0 + nb * 16 + lhalf;
            if (gc >= cout) continue;
#pragma unroll
            for (int r = 0; r < 8; ++r) {
                const int gr = row0 + wave * 32 + ms * 16 + r + (hsel ? 8 : 0);
                if (gr >= rows) continue;
                float y = d.f[r] + bias[gc];
                if (mode == 0) {
                    y = y * (gamma[gc] * bnrs) + beta[gc];
                    y = fmaxf(y, 0.0f);
                }
                Y[(size_t)gr * cout + gc] = y;
            }
        }
    }
}

// ---------------------------------------------------------------------------
// Farthest point sampling: one 1024-thread block per batch, dist in LDS.
// ---------------------------------------------------------------------------
__global__ __launch_bounds__(1024) void fps_kernel(
    const float* __restrict__ xyz, int N, int npoint, int* __restrict__ idx_out)
{
    __shared__ float dd[NPTS];
    __shared__ float sv[1024];
    __shared__ int   si[1024];
    __shared__ int   s_far;

    const int b = blockIdx.x, tid = threadIdx.x;
    const float* px = xyz + (size_t)b * N * 3;

    for (int n = tid; n < N; n += 1024) dd[n] = 1e10f;
    if (tid == 0) s_far = 0;
    __syncthreads();

    for (int it = 0; it < npoint; ++it) {
        const int far = s_far;
        if (tid == 0) idx_out[b * npoint + it] = far;
        const float cx = px[far * 3 + 0], cy = px[far * 3 + 1], cz = px[far * 3 + 2];
        float bv = -1.0f; int bi = 0;
        for (int n = tid; n < N; n += 1024) {
            float dx = px[n * 3 + 0] - cx;
            float dy = px[n * 3 + 1] - cy;
            float dz = px[n * 3 + 2] - cz;
            float d = dx * dx + dy * dy + dz * dz;
            float old = dd[n];
            float nd = old < d ? old : d;
            dd[n] = nd;
            if (nd > bv) { bv = nd; bi = n; }
        }
        sv[tid] = bv; si[tid] = bi;
        __syncthreads();
        for (int s = 512; s > 0; s >>= 1) {
            if (tid < s) {
                if (sv[tid + s] > sv[tid] ||
                    (sv[tid + s] == sv[tid] && si[tid + s] < si[tid])) {
                    sv[tid] = sv[tid + s]; si[tid] = si[tid + s];
                }
            }
            __syncthreads();
        }
        if (tid == 0) s_far = si[0];
        __syncthreads();
    }
}

// ---------------------------------------------------------------------------
// Gather C channels: dst[b,m,c] = src[b, idx[b,m], c]
// ---------------------------------------------------------------------------
__global__ void gather_kernel(const float* __restrict__ src, const int* __restrict__ idx,
                              float* __restrict__ dst, int Nin, int M, int C, int total)
{
    int i = blockIdx.x * blockDim.x + threadIdx.x;
    if (i >= total) return;
    int c = i % C, m = (i / C) % M, b = i / (C * M);
    int g = idx[b * M + m];
    dst[i] = src[((size_t)b * Nin + g) * C + c];
}

// ---------------------------------------------------------------------------
// Ball query: first K indices (ascending) within radius; pad with first.
// ---------------------------------------------------------------------------
__global__ void ball_query_kernel(const float* __restrict__ xyz, const float* __restrict__ nxyz,
                                  int N, int M, int K, float r2,
                                  int* __restrict__ gidx, int total)
{
    int i = blockIdx.x * blockDim.x + threadIdx.x;
    if (i >= total) return; // total = B*M
    int b = i / M;
    const float* px = xyz + (size_t)b * N * 3;
    float cx = nxyz[i * 3 + 0], cy = nxyz[i * 3 + 1], cz = nxyz[i * 3 + 2];
    int* out = gidx + (size_t)i * K;
    int cnt = 0, first = N - 1;
    for (int n = 0; n < N && cnt < K; ++n) {
        float dx = px[n * 3 + 0] - cx;
        float dy = px[n * 3 + 1] - cy;
        float dz = px[n * 3 + 2] - cz;
        float d = dx * dx + dy * dy + dz * dz;
        if (d <= r2) {
            if (cnt == 0) first = n;
            out[cnt++] = n;
        }
    }
    for (int k = cnt; k < K; ++k) out[k] = first;
}

// ---------------------------------------------------------------------------
// Grouping: feat[b,m,k,:] = [xyz[g]-nxyz[m] (3), pts[g] (C)], padded stride Ctp
// ---------------------------------------------------------------------------
__global__ void group_kernel(const float* __restrict__ xyz, const float* __restrict__ pts,
                             const float* __restrict__ nxyz, const int* __restrict__ gidx,
                             float* __restrict__ feat, int N, int M, int K, int C,
                             int Ctp, int total)
{
    int i = blockIdx.x * blockDim.x + threadIdx.x;
    if (i >= total) return; // total = B*M*K*(3+C)
    int Ct = 3 + C;
    int c = i % Ct;
    int k = (i / Ct) % K;
    int m = (i / (Ct * K)) % M;
    int b = i / (Ct * K * M);
    int g = gidx[((size_t)b * M + m) * K + k];
    float v;
    if (c < 3) v = xyz[((size_t)b * N + g) * 3 + c] - nxyz[((size_t)b * M + m) * 3 + c];
    else       v = pts[((size_t)b * N + g) * C + (c - 3)];
    feat[((size_t)(b * M + m) * K + k) * Ctp + c] = v;
}

// ---------------------------------------------------------------------------
// Max-pool over K, write to concat destination at channel offset.
// ---------------------------------------------------------------------------
__global__ void maxpool_kernel(const float* __restrict__ X, float* __restrict__ Y,
                               int K, int C, int Cout, int coff, int total)
{
    int i = blockIdx.x * blockDim.x + threadIdx.x;
    if (i >= total) return; // total = B*M*C
    int c = i % C, bm = i / C;
    const float* xp = X + (size_t)bm * K * C + c;
    float v = xp[0];
    for (int k = 1; k < K; ++k) v = fmaxf(v, xp[(size_t)k * C]);
    Y[(size_t)bm * Cout + coff + c] = v;
}

// ---------------------------------------------------------------------------
// Copy channels into a wider concat destination.
// ---------------------------------------------------------------------------
__global__ void copy_ch_kernel(const float* __restrict__ src, float* __restrict__ dst,
                               int Csrc, int Cdst, int coff, int total)
{
    int i = blockIdx.x * blockDim.x + threadIdx.x;
    if (i >= total) return; // total = rows*Csrc
    int c = i % Csrc, r = i / Csrc;
    dst[(size_t)r * Cdst + coff + c] = src[i];
}

// ---------------------------------------------------------------------------
// 3-NN inverse-distance interpolation; xyz2 staged in LDS when possible.
// ---------------------------------------------------------------------------
__global__ __launch_bounds__(256) void three_interp_kernel(
    const float* __restrict__ x1, const float* __restrict__ x2,
    const float* __restrict__ p2, float* __restrict__ dst,
    int N1, int N2, int C2, int Cdst, int coff, int total)
{
    __shared__ float qs[1024 * 3];
    const int tid = threadIdx.x;
    const int base = blockIdx.x * 256;
    const int b0 = base / N1;
    const int b1 = (base + 255) / N1;
    const bool uni = (b0 == b1) && (N2 <= 1024);
    if (uni) {
        for (int t = tid; t < N2 * 3; t += 256) qs[t] = x2[(size_t)b0 * N2 * 3 + t];
    }
    __syncthreads();

    const int i = base + tid;
    if (i >= total) return; // total = B*N1
    const int b = i / N1;
    float ax = x1[i * 3 + 0], ay = x1[i * 3 + 1], az = x1[i * 3 + 2];
    const float* q = uni ? qs : (x2 + (size_t)b * N2 * 3);
    float d0 = 1e30f, d1 = 1e30f, d2v = 1e30f;
    int i0 = 0, i1 = 0, i2 = 0;
    for (int j = 0; j < N2; ++j) {
        float dx = q[j * 3 + 0] - ax;
        float dy = q[j * 3 + 1] - ay;
        float dz = q[j * 3 + 2] - az;
        float d = dx * dx + dy * dy + dz * dz;
        if (d < d0)       { d2v = d1; i2 = i1; d1 = d0; i1 = i0; d0 = d; i0 = j; }
        else if (d < d1)  { d2v = d1; i2 = i1; d1 = d;  i1 = j; }
        else if (d < d2v) { d2v = d;  i2 = j; }
    }
    float w0 = 1.0f / (d0 + 1e-8f);
    float w1 = 1.0f / (d1 + 1e-8f);
    float w2 = 1.0f / (d2v + 1e-8f);
    float ws = w0 + w1 + w2;
    w0 /= ws; w1 /= ws; w2 /= ws;
    const float* pb = p2 + (size_t)b * N2 * C2;
    float* o = dst + (size_t)i * Cdst + coff;
    for (int c = 0; c < C2; ++c)
        o[c] = w0 * pb[(size_t)i0 * C2 + c] + w1 * pb[(size_t)i1 * C2 + c]
             + w2 * pb[(size_t)i2 * C2 + c];
}

// ---------------------------------------------------------------------------
// Split input_pc [B,N,6] -> xyz [B,N,3], feats [B,N,3]
// ---------------------------------------------------------------------------
__global__ void split_kernel(const float* __restrict__ in, float* __restrict__ xyz,
                             float* __restrict__ f, int total)
{
    int i = blockIdx.x * blockDim.x + threadIdx.x;
    if (i >= total) return; // total = B*N
    xyz[i * 3 + 0] = in[i * 6 + 0];
    xyz[i * 3 + 1] = in[i * 6 + 1];
    xyz[i * 3 + 2] = in[i * 6 + 2];
    f[i * 3 + 0]   = in[i * 6 + 3];
    f[i * 3 + 1]   = in[i * 6 + 4];
    f[i * 3 + 2]   = in[i * 6 + 5];
}

// ---------------------------------------------------------------------------
// Host-side orchestration
// ---------------------------------------------------------------------------
struct Layer { const float *w, *b, *g, *be; int cin, cout; };

static void take_layer(void* const* d_in, int& p, Layer* L, int cin, int cout)
{
    // insertion order in reference: w, b, g, be
    L->w  = (const float*)d_in[p++];
    L->b  = (const float*)d_in[p++];
    L->g  = (const float*)d_in[p++];
    L->be = (const float*)d_in[p++];
    L->cin = cin; L->cout = cout;
}

static void build_chain(void* const* d_in, int& p, Layer* L, int cin,
                        const int* chans, int n)
{
    for (int i = 0; i < n; ++i) { take_layer(d_in, p, &L[i], cin, chans[i]); cin = chans[i]; }
}

static float* run_mlp(hipStream_t st, const Layer* L, int nl, int rows, int ldx0,
                      const float* in, float* t0, float* t1, float* fin)
{
    const float* cur = in;
    float* outp = nullptr;
    int ldx = ldx0;
    for (int i = 0; i < nl; ++i) {
        float* dst = (i == nl - 1 && fin) ? fin : ((cur == t0) ? t1 : t0);
        dim3 g(ceil_div_i(rows, TM), ceil_div_i(L[i].cout, TN));
        gemm_bn_relu_kernel<<<g, 256, 0, st>>>(cur, ldx, L[i].w, L[i].b, L[i].g, L[i].be,
                                               dst, rows, L[i].cin, L[i].cout, 0);
        cur = dst; outp = dst; ldx = L[i].cout;
    }
    return outp;
}

static void run_sa(hipStream_t st, const float* xyz_in, int Nin,
                   const float* pts_in, int Cpts, int npoint,
                   const float* radii, const int* ks, Layer* s0, Layer* s1,
                   int* fps_idx, int* gidx,
                   float* featbuf, float* t0, float* t1,
                   float* nxyz_out, float* cat_out)
{
    fps_kernel<<<BATCH, 1024, 0, st>>>(xyz_in, Nin, npoint, fps_idx);
    int tot = BATCH * npoint * 3;
    gather_kernel<<<ceil_div_i(tot, 256), 256, 0, st>>>(xyz_in, fps_idx, nxyz_out,
                                                        Nin, npoint, 3, tot);
    const int Cout_tot = s0[2].cout + s1[2].cout;
    int coff = 0;
    for (int sidx = 0; sidx < 2; ++sidx) {
        Layer* L = sidx ? s1 : s0;
        const int K = ks[sidx];
        const float r = radii[sidx];
        int totq = BATCH * npoint;
        ball_query_kernel<<<ceil_div_i(totq, 256), 256, 0, st>>>(
            xyz_in, nxyz_out, Nin, npoint, K, r * r, gidx, totq);
        const int Ct = 3 + Cpts, Ctp = pad4(Ct);
        int totg = BATCH * npoint * K * Ct;
        group_kernel<<<ceil_div_i(totg, 256), 256, 0, st>>>(
            xyz_in, pts_in, nxyz_out, gidx, featbuf, Nin, npoint, K, Cpts, Ctp, totg);
        int rows = BATCH * npoint * K;
        float* mo = run_mlp(st, L, 3, rows, Ctp, featbuf, t0, t1, nullptr);
        int totp = BATCH * npoint * L[2].cout;
        maxpool_kernel<<<ceil_div_i(totp, 256), 256, 0, st>>>(
            mo, cat_out, K, L[2].cout, Cout_tot, coff, totp);
        coff += L[2].cout;
    }
}

static void run_fp(hipStream_t st, const float* x1, int N1, const float* x2, int N2,
                   const float* p1, int C1, const float* p2, int C2,
                   Layer* L, int nl, float* featbuf, float* t0, float* t1, float* dst)
{
    const int Cin = C1 + C2, Cinp = pad4(Cin);
    if (p1) {
        int tot = BATCH * N1 * C1;
        copy_ch_kernel<<<ceil_div_i(tot, 256), 256, 0, st>>>(p1, featbuf, C1, Cinp, 0, tot);
    }
    int tot = BATCH * N1;
    three_interp_kernel<<<ceil_div_i(tot, 256), 256, 0, st>>>(
        x1, x2, p2, featbuf, N1, N2, C2, Cinp, C1, tot);
    run_mlp(st, L, nl, BATCH * N1, Cinp, featbuf, t0, t1, dst);
}

extern "C" void kernel_launch(void* const* d_in, const int* in_sizes, int n_in,
                              void* d_out, int out_size, void* d_ws, size_t ws_size,
                              hipStream_t stream)
{
    if (n_in < 139) return;

    // ---- parse parameters (setup_inputs dict insertion order) ----
    int p = 1; // d_in[0] = input_pc
    Layer sa1a[3], sa1b[3], sa2a[3], sa2b[3], sa3a[3], sa3b[3], sa4a[3], sa4b[3];
    Layer fp4L[2], fp3L[2], fp2L[2], fp1L[3], head1L[1];
    const int c1a[3] = {32, 32, 64},  c1b[3] = {64, 64, 128};
    const int c4a[3] = {128, 256, 512};
    build_chain(d_in, p, sa1a, 6,   c1a, 3);
    build_chain(d_in, p, sa1b, 6,   c1b, 3);
    build_chain(d_in, p, sa2a, 195, c1a, 3);
    build_chain(d_in, p, sa2b, 195, c1b, 3);
    build_chain(d_in, p, sa3a, 195, c1b, 3);
    build_chain(d_in, p, sa3b, 195, c1b, 3);
    build_chain(d_in, p, sa4a, 259, c4a, 3);
    build_chain(d_in, p, sa4b, 259, c4a, 3);
    const int cfp4[2] = {256, 256}, cfp3[2] = {256, 256}, cfp2[2] = {256, 128};
    const int cfp1[3] = {128, 128, 128}, ch1[1] = {128};
    build_chain(d_in, p, fp4L, 1280, cfp4, 2);
    build_chain(d_in, p, fp3L, 448,  cfp3, 2);
    build_chain(d_in, p, fp2L, 448,  cfp2, 2);
    build_chain(d_in, p, fp1L, 128,  cfp1, 3);
    build_chain(d_in, p, head1L, 128, ch1, 1);
    const float* h2w = (const float*)d_in[p++]; // [128,13]
    const float* h2b = (const float*)d_in[p++]; // [13]

    // ---- workspace bump allocator ----
    char* base = (char*)d_ws;
    size_t off = 0;
    auto alloc_f = [&](size_t nfloats) -> float* {
        float* r = (float*)(base + off);
        off += ((nfloats * sizeof(float)) + 255) & ~(size_t)255;
        return r;
    };
    auto alloc_i = [&](size_t nints) -> int* {
        int* r = (int*)(base + off);
        off += ((nints * sizeof(int)) + 255) & ~(size_t)255;
        return r;
    };

    float* xyz0 = alloc_f((size_t)BATCH * NPTS * 3);
    float* f0   = alloc_f((size_t)BATCH * NPTS * 3);
    float* l1x  = alloc_f((size_t)BATCH * 1024 * 3);
    float* l1p  = alloc_f((size_t)BATCH * 1024 * 192);
    float* l2x  = alloc_f((size_t)BATCH * 512 * 3);
    float* l2p  = alloc_f((size_t)BATCH * 512 * 192);
    float* l3x  = alloc_f((size_t)BATCH * 256 * 3);
    float* l3p  = alloc_f((size_t)BATCH * 256 * 256);
    float* l4x  = alloc_f((size_t)BATCH * 128 * 3);
    float* l4p  = alloc_f((size_t)BATCH * 128 * 1024);
    float* l3pn = alloc_f((size_t)BATCH * 256 * 256);
    float* l2pn = alloc_f((size_t)BATCH * 512 * 256);
    float* l1pn = alloc_f((size_t)BATCH * 1024 * 128);
    float* l0p  = alloc_f((size_t)BATCH * NPTS * 128);
    int*   fps_idx = alloc_i((size_t)BATCH * 1024);
    int*   gidx    = alloc_i((size_t)BATCH * 1024 * 32);
    float* featbuf = alloc_f((size_t)26000000);  // max grouped feat: 131072*196
    float* t0      = alloc_f((size_t)33554432);  // max GEMM out: 262144*128
    float* t1      = alloc_f((size_t)33554432);
    (void)ws_size;

    // ---- forward pass ----
    int tot = BATCH * NPTS;
    split_kernel<<<ceil_div_i(tot, 256), 256, 0, stream>>>(
        (const float*)d_in[0], xyz0, f0, tot);

    const float r1[2] = {1.0f, 3.0f}; const int k1[2] = {8, 32};
    const float r2[2] = {2.0f, 4.0f}; const int k2[2] = {8, 32};
    const float r3[2] = {3.0f, 6.0f}; const int k3[2] = {16, 32};
    const float r4[2] = {4.0f, 8.0f}; const int k4[2] = {16, 32};

    run_sa(stream, xyz0, NPTS, f0,  3,   1024, r1, k1, sa1a, sa1b,
           fps_idx, gidx, featbuf, t0, t1, l1x, l1p);
    run_sa(stream, l1x, 1024, l1p, 192, 512,  r2, k2, sa2a, sa2b,
           fps_idx, gidx, featbuf, t0, t1, l2x, l2p);
    run_sa(stream, l2x, 512,  l2p, 192, 256,  r3, k3, sa3a, sa3b,
           fps_idx, gidx, featbuf, t0, t1, l3x, l3p);
    run_sa(stream, l3x, 256,  l3p, 256, 128,  r4, k4, sa4a, sa4b,
           fps_idx, gidx, featbuf, t0, t1, l4x, l4p);

    run_fp(stream, l3x, 256,  l4x, 128, l3p, 256, l4p,  1024, fp4L, 2,
           featbuf, t0, t1, l3pn);
    run_fp(stream, l2x, 512,  l3x, 256, l2p, 192, l3pn, 256,  fp3L, 2,
           featbuf, t0, t1, l2pn);
    run_fp(stream, l1x, 1024, l2x, 512, l1p, 192, l2pn, 256,  fp2L, 2,
           featbuf, t0, t1, l1pn);
    run_fp(stream, xyz0, NPTS, l1x, 1024, nullptr, 0, l1pn, 128, fp1L, 3,
           featbuf, t0, t1, l0p);

    // head: h = bn-relu(l0p @ head1); out = h @ head2.w + head2.b
    const int rows = BATCH * NPTS;
    {
        dim3 g(ceil_div_i(rows, TM), ceil_div_i(128, TN));
        gemm_bn_relu_kernel<<<g, 256, 0, stream>>>(
            l0p, 128, head1L[0].w, head1L[0].b, head1L[0].g, head1L[0].be,
            t0, rows, 128, 128, 0);
    }
    {
        dim3 g(ceil_div_i(rows, TM), ceil_div_i(13, TN));
        gemm_bn_relu_kernel<<<g, 256, 0, stream>>>(
            t0, 128, h2w, h2b, h2b, h2b, (float*)d_out, rows, 128, 13, 1);
    }
    (void)in_sizes; (void)out_size;
}
